// LstmCell_13030930776239
// MI455X (gfx1250) — matrix operand order repaired and checked by reading, not compiled, not run on hardware
//
#include <hip/hip_runtime.h>

// ---------------------------------------------------------------------------
// LSTM cell for MI455X (gfx1250, wave32, WMMA + Tensor Data Mover).
//   ifgo = [X|H] @ [Wi;Wh] + (b_i+b_h)   (B=4096, I=H=1024, K=2048, N=4096)
//   i,f,o = sigmoid; g = tanh; c' = f*c + i*g; h' = o*tanh(c')
// Round 4: widen per-wave M to 32 rows (2 A fragments) so each B fragment
// ds_load feeds two back-to-back WMMAs: 16 WMMA / 20 ds_load per K-step
// (was 8 / 18), halving LDS bandwidth per FLOP. TDM double-buffer pipeline,
// pad_enable bank-deconfliction, and the smem escape/clobber all retained.
// ---------------------------------------------------------------------------

typedef __bf16 bf16_t;
typedef __attribute__((ext_vector_type(16))) __bf16 v16bf;
typedef __attribute__((ext_vector_type(8)))  __bf16 v8bf;
typedef __attribute__((ext_vector_type(8)))  float  v8f;
typedef __attribute__((ext_vector_type(4)))  unsigned int u32x4;
typedef __attribute__((ext_vector_type(8)))  int    i32x8;
typedef __attribute__((ext_vector_type(4)))  int    i32x4;

#define LSTM_B 4096
#define LSTM_H 1024
#define LSTM_K 2048              // I + H
#define LSTM_N 4096              // 4*H

// workspace layout (bytes)
#define WS_A_OFF    0                            // bf16 [4096][2048]  = 16 MiB
#define WS_W_OFF    (16u * 1024u * 1024u)        // bf16 [2048][4096]  = 16 MiB
#define WS_BIAS_OFF (32u * 1024u * 1024u)        // f32  [4096]

// LDS geometry: rows are 32 bf16 (64B) + 16B TDM pad -> 40-element stride.
#define ROW_ELEMS   40u
#define BLK_M       256u                         // batch rows per block
#define A_TILE_B    (BLK_M * 80u)                // 20480 B
#define W_TILE_B    (128u * 80u)                 // 10240 B (4 gates x 32 k-rows)
#define LDS_BUF_BYTES (A_TILE_B + W_TILE_B)      // 30720 B per buffer

static __device__ __forceinline__ unsigned short f32_to_bf16(float f) {
    unsigned u = __float_as_uint(f);
    unsigned r = (u + 0x7FFFu + ((u >> 16) & 1u)) >> 16;   // round-nearest-even
    return (unsigned short)r;
}
static __device__ __forceinline__ unsigned pack2(float a, float b) {
    return (unsigned)f32_to_bf16(a) | ((unsigned)f32_to_bf16(b) << 16);
}
static __device__ __forceinline__ float sigmoidf_(float x) {
    return 1.0f / (1.0f + __expf(-x));
}

// ---- TDM descriptor build + issue (ISA cdna5 §8.3-8.6) --------------------
// Loads a (tile_d0 x tile_d1 x tile_d2) tile of 2-byte elements:
//   global addr = gaddr + 2*(x + y*stride0 + z*stride1)
//   LDS fill: x fastest, contiguous from lds_off, +16B pad after every 64B.
static __device__ __forceinline__ void tdm_load_tile(
        unsigned lds_off, const void* gaddr,
        unsigned tile_d0, unsigned tile_d1, unsigned tile_d2,
        unsigned long long stride0, unsigned long long stride1) {
    unsigned long long ga = (unsigned long long)gaddr;

    u32x4 g0;
    g0[0] = 1u;                                         // count=1, user mode
    g0[1] = lds_off;                                    // lds_addr (bytes)
    g0[2] = (unsigned)(ga & 0xFFFFFFFFull);             // global_addr[31:0]
    g0[3] = (unsigned)((ga >> 32) & 0x1FFFFFFull)       // global_addr[56:32]
          | 0x80000000u;                                // type=2 ("image")

    i32x8 g1;
    g1[0] = (int)((1u << 16)                            // data_size=1 (2 bytes)
          |       (1u << 20)                            // pad_enable
          |       (3u << 22)                            // pad_interval: 16 DW (64B)
          |       (3u << 25));                          // pad_amount: 4 DW (16B)
    g1[1] = (int)(tile_d0 << 16);                       // tensor_dim0[15:0]=tile_d0
    g1[2] = (int)(tile_d1 << 16);                       // dim0 hi=0 | tensor_dim1 lo
    g1[3] = (int)(tile_d0 << 16);                       // dim1 hi=0 | tile_dim0
    g1[4] = (int)(tile_d1 | (tile_d2 << 16));           // tile_dim1 | tile_dim2
    g1[5] = (int)(unsigned)(stride0 & 0xFFFFFFFFull);   // tensor_dim0_stride lo
    g1[6] = (int)(((stride0 >> 32) & 0xFFFFull)
          | ((stride1 & 0xFFFFull) << 16));             // stride0 hi | stride1 lo
    g1[7] = (int)((stride1 >> 16) & 0xFFFFFFFFull);     // stride1[47:16]

    i32x4 g2;
    g2[0] = (int)(tile_d2 ? tile_d2 : 1u);              // tensor_dim2
    g2[1] = 0; g2[2] = 0; g2[3] = 0;
    i32x4 g3;
    g3[0] = 0; g3[1] = 0; g3[2] = 0; g3[3] = 0;

#if defined(__clang_major__) && (__clang_major__ >= 23)
    i32x8 g4;
    g4[0] = 0; g4[1] = 0; g4[2] = 0; g4[3] = 0;
    g4[4] = 0; g4[5] = 0; g4[6] = 0; g4[7] = 0;
    __builtin_amdgcn_tensor_load_to_lds(g0, g1, g2, g3, g4, 0);
#else
    __builtin_amdgcn_tensor_load_to_lds(g0, g1, g2, g3, 0);
#endif
}

// --------------------------- convert kernels -------------------------------

// A[b][k] = (k<1024) ? X[b][k] : H[b][k-1024], bf16. 8 elems / thread.
__global__ void lstm_pack_A(const float* __restrict__ x,
                            const float* __restrict__ h,
                            unsigned short* __restrict__ A) {
    unsigned idx = blockIdx.x * 256u + threadIdx.x;      // 1,048,576 total
    unsigned row = idx >> 8;
    unsigned k   = (idx & 255u) * 8u;
    const float* src = (k < 1024u) ? (x + row * 1024u + k)
                                   : (h + row * 1024u + (k - 1024u));
    float4 f0 = *(const float4*)(src);
    float4 f1 = *(const float4*)(src + 4);
    uint4 o;
    o.x = pack2(f0.x, f0.y); o.y = pack2(f0.z, f0.w);
    o.z = pack2(f1.x, f1.y); o.w = pack2(f1.z, f1.w);
    *(uint4*)(A + row * 2048u + k) = o;
}

// W[k][n] = (k<1024) ? Wi[k][n] : Wh[k-1024][n], bf16 row-major.
__global__ void lstm_pack_W(const float* __restrict__ wi,
                            const float* __restrict__ wh,
                            unsigned short* __restrict__ W) {
    unsigned idx = blockIdx.x * 256u + threadIdx.x;      // 1,048,576 total
    unsigned row = idx >> 9;
    unsigned n   = (idx & 511u) * 8u;
    const float* src = (row < 1024u) ? (wi + row * 4096u + n)
                                     : (wh + (row - 1024u) * 4096u + n);
    float4 f0 = *(const float4*)(src);
    float4 f1 = *(const float4*)(src + 4);
    uint4 o;
    o.x = pack2(f0.x, f0.y); o.y = pack2(f0.z, f0.w);
    o.z = pack2(f1.x, f1.y); o.w = pack2(f1.z, f1.w);
    *(uint4*)(W + row * 4096u + n) = o;
}

__global__ void lstm_pack_bias(const float* __restrict__ bi,
                               const float* __restrict__ bh,
                               float* __restrict__ bias) {
    unsigned idx = blockIdx.x * 256u + threadIdx.x;
    if (idx < 4096u) bias[idx] = bi[idx] + bh[idx];
}

// --------------------------- fused GEMM + gates ----------------------------
// grid: (LSTM_H/32, LSTM_B/256) ; block: 256 threads = 8 waves (wave32).
// Tile: 256 batch rows x 32 n-cols for all 4 gates; each wave owns 32 rows
// (2 A fragments). Double-buffered LDS; TDM DMA for kk+32 overlaps WMMA on kk.
__global__ void __launch_bounds__(256)
lstm_wmma_kernel(const unsigned short* __restrict__ Abf,
                 const unsigned short* __restrict__ Wbf,
                 const float* __restrict__ bias,
                 const float* __restrict__ c_prev,
                 float* __restrict__ out) {
    __shared__ __align__(128) char smem[2 * LDS_BUF_BYTES];   // 61440 B

    const unsigned tid  = threadIdx.x;
    const unsigned wave = tid >> 5;
    const unsigned lane = tid & 31u;
    const unsigned half = lane >> 4;          // 0: lanes 0-15, 1: lanes 16-31
    const unsigned l16  = lane & 15u;

    const unsigned n0 = blockIdx.x * 32u;     // n within a gate [0,1024)
    const unsigned b0 = blockIdx.y * BLK_M;   // batch row base

    const v8f vzero = {0.f, 0.f, 0.f, 0.f, 0.f, 0.f, 0.f, 0.f};
    v8f acc[4][2][2];                         // [gate][n-subtile][m-frag]
#pragma unroll
    for (int g = 0; g < 4; ++g)
#pragma unroll
        for (int t = 0; t < 2; ++t)
#pragma unroll
            for (int m = 0; m < 2; ++m) acc[g][t][m] = vzero;

    // ---- prologue: DMA first K tile into buffer 0 -------------------------
    if (tid < 32u) {
        // A tile: x=k (32 contiguous), y=row (256, stride 2048)
        tdm_load_tile(0u, &Abf[(size_t)b0 * 2048u],
                      32u, BLK_M, 0u, 2048ull, 0ull);
        // W tile: x=n (32 contiguous), y=k (32, stride 4096), z=gate (4, stride 1024)
        tdm_load_tile(A_TILE_B, &Wbf[n0],
                      32u, 32u, 4u, 4096ull, 1024ull);
        __builtin_amdgcn_s_wait_tensorcnt(0);
    }
    __syncthreads();

    unsigned cur = 0;
    for (unsigned kk = 0; kk < LSTM_K; kk += 32) {
        const unsigned nxt = cur ^ 1u;
        const bool more = (kk + 32u) < LSTM_K;

        // ---- issue DMA for next K tile into the other buffer --------------
        if (more && tid < 32u) {
            tdm_load_tile(nxt * LDS_BUF_BYTES,
                          &Abf[(size_t)b0 * 2048u + kk + 32u],
                          32u, BLK_M, 0u, 2048ull, 0ull);
            tdm_load_tile(nxt * LDS_BUF_BYTES + A_TILE_B,
                          &Wbf[(size_t)(kk + 32u) * 4096u + n0],
                          32u, 32u, 4u, 4096ull, 1024ull);
        }

        // TDM isn't modeled as writing `smem`: escape it into an opaque asm
        // with a memory clobber so the ds_loads below aren't folded away.
        asm volatile("" : : "v"(&smem[0]) : "memory");

        // ---- compute on current buffer ------------------------------------
        const bf16_t* lA = (const bf16_t*)(smem + cur * LDS_BUF_BYTES);
        const bf16_t* lW = (const bf16_t*)(smem + cur * LDS_BUF_BYTES + A_TILE_B);

        // Two A fragments: lane holds M = wave*32 + m*16 + l16,
        // two 8-wide K chunks selected by lane half.
        v16bf af[2];
#pragma unroll
        for (int m = 0; m < 2; ++m) {
            const unsigned arow = (wave * 32u + (unsigned)m * 16u + l16) * ROW_ELEMS;
            v8bf alo = *(const v8bf*)&lA[arow + half * 8u];
            v8bf ahi = *(const v8bf*)&lA[arow + 16u + half * 8u];
            af[m] = __builtin_shufflevector(alo, ahi,
                0, 1, 2, 3, 4, 5, 6, 7, 8, 9, 10, 11, 12, 13, 14, 15);
        }

        // Each B fragment feeds two WMMAs (m=0,1): 16 WMMA per K-step.
        const unsigned bcol = l16;  // n-col within 16-wide subtile
#pragma unroll
        for (int g = 0; g < 4; ++g) {
#pragma unroll
            for (int t = 0; t < 2; ++t) {
                v16bf bfrag;
#pragma unroll
                for (int s = 0; s < 16; ++s) {
                    const unsigned bk = (s < 8) ? (half * 8u + (unsigned)s)
                                                : (16u + half * 8u + (unsigned)(s - 8));
                    bfrag[s] = lW[(unsigned)g * (32u * ROW_ELEMS) + bk * ROW_ELEMS
                                  + (unsigned)t * 16u + bcol];
                }
#pragma unroll
                for (int m = 0; m < 2; ++m)
                    acc[g][t][m] = __builtin_amdgcn_wmma_f32_16x16x32_bf16(
                        false, af[m], false, bfrag, (short)0,
                        acc[g][t][m], false, false);
            }
        }

        // ---- drain DMA, publish next buffer -------------------------------
        if (more && tid < 32u) {
            __builtin_amdgcn_s_wait_tensorcnt(0);
        }
        __syncthreads();
        cur = nxt;
    }

    // ---- epilogue: bias + activations + cell update -----------------------
#pragma unroll
    for (int t = 0; t < 2; ++t) {
        const unsigned n  = n0 + (unsigned)t * 16u + l16;   // [0,1024)
        const float bI = bias[n];
        const float bF = bias[1024u + n];
        const float bG = bias[2048u + n];
        const float bO = bias[3072u + n];
#pragma unroll
        for (int m = 0; m < 2; ++m) {
#pragma unroll
            for (int r = 0; r < 8; ++r) {
                const unsigned b = b0 + wave * 32u + (unsigned)m * 16u
                                 + half * 8u + (unsigned)r;
                const float iv = sigmoidf_(acc[0][t][m][r] + bI);
                const float fv = sigmoidf_(acc[1][t][m][r] + bF);
                const float gv = tanhf(acc[2][t][m][r] + bG);
                const float ov = sigmoidf_(acc[3][t][m][r] + bO);
                const float c_old = c_prev[b * 1024u + n];
                const float c_new = fv * c_old + iv * gv;
                const float h_new = ov * tanhf(c_new);
                out[b * 1024u + n] = h_new;                                // h_new
                out[(size_t)LSTM_B * LSTM_H + b * 1024u + n] = c_new;      // c_new
            }
        }
    }
}

// ------------------------------ launcher -----------------------------------
extern "C" void kernel_launch(void* const* d_in, const int* in_sizes, int n_in,
                              void* d_out, int out_size, void* d_ws, size_t ws_size,
                              hipStream_t stream) {
    const float* x    = (const float*)d_in[0];
    const float* h_t  = (const float*)d_in[1];
    const float* c_t  = (const float*)d_in[2];
    const float* w_i  = (const float*)d_in[3];
    const float* w_h  = (const float*)d_in[4];
    const float* b_i  = (const float*)d_in[5];
    const float* b_h  = (const float*)d_in[6];

    char* ws = (char*)d_ws;
    unsigned short* A    = (unsigned short*)(ws + WS_A_OFF);
    unsigned short* W    = (unsigned short*)(ws + WS_W_OFF);
    float*          bias = (float*)(ws + WS_BIAS_OFF);

    lstm_pack_A   <<<4096, 256, 0, stream>>>(x, h_t, A);
    lstm_pack_W   <<<4096, 256, 0, stream>>>(w_i, w_h, W);
    lstm_pack_bias<<<16,   256, 0, stream>>>(b_i, b_h, bias);

    dim3 grid(LSTM_H / 32, LSTM_B / BLK_M);   // 32 x 16 workgroups
    lstm_wmma_kernel<<<grid, 256, 0, stream>>>(A, W, bias, c_t, (float*)d_out);
}
